// MultiHeadAttention_3298534883852
// MI455X (gfx1250) — matrix operand order repaired and checked by reading, not compiled, not run on hardware
//
#include <hip/hip_runtime.h>
#include <hip/hip_bf16.h>

typedef __attribute__((ext_vector_type(16))) __bf16 v16bf;
typedef __attribute__((ext_vector_type(8)))  __bf16 v8bf;
typedef __attribute__((ext_vector_type(8)))  float  v8f;
typedef __attribute__((ext_vector_type(4)))  int    v4i;

#define HID    2048
#define NHEADS 16
#define NKV    4
#define HD     128
#define BATCH  2
#define SEQ    2048
#define ROWS   (BATCH * SEQ)      // 4096
#define QKV_N  3072               // 2048 q + 512 k + 512 v

// ---- CDNA5 async global->LDS path (guarded; sync fallback keeps compile green)
#if defined(__has_builtin)
#  if __has_builtin(__builtin_amdgcn_global_load_async_to_lds_b128) && \
      __has_builtin(__builtin_amdgcn_s_wait_asynccnt)
#    define HAS_ASYNC_LDS 1
#  endif
#endif
#ifndef HAS_ASYNC_LDS
#  define HAS_ASYNC_LDS 0
#endif
#define GAS __attribute__((address_space(1)))
#define LAS __attribute__((address_space(3)))

__device__ __forceinline__ void cp16_g2l(__bf16* l, const __bf16* g) {
#if HAS_ASYNC_LDS
    __builtin_amdgcn_global_load_async_to_lds_b128((GAS v4i*)(g), (LAS v4i*)(l), 0, 0);
#else
    *(v8bf*)l = *(const v8bf*)g;
#endif
}
__device__ __forceinline__ void async_join() {
#if HAS_ASYNC_LDS
    __builtin_amdgcn_s_wait_asynccnt(0);
#endif
}

__device__ __forceinline__ __bf16 f2bf(float f) {
    unsigned u = __builtin_bit_cast(unsigned, f);
    u += 0x7FFFu + ((u >> 16) & 1u);            // round-to-nearest-even
    unsigned short h = (unsigned short)(u >> 16);
    return __builtin_bit_cast(__bf16, h);
}

__device__ __forceinline__ v8f zero8() {
    v8f z;
#pragma unroll
    for (int i = 0; i < 8; ++i) z[i] = 0.0f;
    return z;
}

// A fragment (16x32 bf16, M x K): lane l -> row m0 + (l&15);
// element i -> K = i + 16*half (contiguous).
__device__ __forceinline__ v16bf load_a_frag(const __bf16* A, int lda, int m0, int k0) {
    const int lane = threadIdx.x & 31;
    const int half = lane >> 4;
    const __bf16* p = A + (size_t)(m0 + (lane & 15)) * lda + k0 + 16 * half;
    v8bf lo = *(const v8bf*)(p);
    v8bf hi = *(const v8bf*)(p + 8);
    return __builtin_shufflevector(lo, hi, 0,1,2,3,4,5,6,7,8,9,10,11,12,13,14,15);
}

// B fragment (32x16 bf16, K x N) from K-major storage Bt[n][k]:
// lane l -> col n0 + (l&15), element i -> K = k0 + i + 16*half  => contiguous 32B.
__device__ __forceinline__ v16bf load_b_frag(const __bf16* Bt, int ldb, int n0, int k0) {
    const int lane = threadIdx.x & 31;
    return *(const v16bf*)(Bt + (size_t)(n0 + (lane & 15)) * ldb + k0 + 16 * (lane >> 4));
}

// ---------------------------------------------------------------- converts
__global__ void f32_to_bf16_kernel(const float* __restrict__ src,
                                   __bf16* __restrict__ dst, long long n) {
    long long i = (long long)blockIdx.x * blockDim.x + threadIdx.x;
    long long stride = (long long)gridDim.x * blockDim.x;
    for (; i < n; i += stride) dst[i] = f2bf(src[i]);
}

// src[K][N] fp32 (row-major) -> dst[N][K] bf16
__global__ void transpose_f32_to_bf16_kernel(const float* __restrict__ src,
                                             __bf16* __restrict__ dst, int K, int N) {
    long long total = (long long)K * N;
    long long i = (long long)blockIdx.x * blockDim.x + threadIdx.x;
    long long stride = (long long)gridDim.x * blockDim.x;
    for (; i < total; i += stride) {
        int k = (int)(i / N), n = (int)(i % N);
        dst[(size_t)n * K + k] = f2bf(src[i]);
    }
}

// ---------------------------------------------------------------- WMMA GEMM
// C[M][N] fp32 = A[M][K] bf16 (row-major) * Bt[N][K] bf16 (K-major).
// 128 threads = 4 waves in 2x2; wave tile 64x64 (4x4 subtiles); block tile 128x128.
__global__ __launch_bounds__(128)
void gemm_bf16_wmma(const __bf16* __restrict__ A, const __bf16* __restrict__ Bt,
                    float* __restrict__ C, int M, int N, int K) {
    const int lane = threadIdx.x & 31;
    const int wid  = threadIdx.x >> 5;
    const int half = lane >> 4;
    const int col  = lane & 15;
    const int m_base = blockIdx.y * 128 + (wid >> 1) * 64;
    const int n_base = blockIdx.x * 128 + (wid & 1) * 64;

    v8f acc[4][4];
#pragma unroll
    for (int i = 0; i < 4; ++i)
#pragma unroll
        for (int j = 0; j < 4; ++j) acc[i][j] = zero8();

    for (int k0 = 0; k0 < K; k0 += 32) {
        v16bf a[4], b[4];
#pragma unroll
        for (int i = 0; i < 4; ++i) a[i] = load_a_frag(A, K, m_base + 16 * i, k0);
#pragma unroll
        for (int j = 0; j < 4; ++j) b[j] = load_b_frag(Bt, K, n_base + 16 * j, k0);
        // deep prefetch of upcoming tiles into cache hierarchy
        __builtin_prefetch(A + (size_t)(m_base + col) * K + k0 + 512, 0, 0);
        __builtin_prefetch(Bt + (size_t)(n_base + col) * K + k0 + 512, 0, 0);
#pragma unroll
        for (int i = 0; i < 4; ++i)
#pragma unroll
            for (int j = 0; j < 4; ++j)
                acc[i][j] = __builtin_amdgcn_wmma_f32_16x16x32_bf16(
                    false, a[i], false, b[j], (short)0, acc[i][j], false, false);
    }

#pragma unroll
    for (int i = 0; i < 4; ++i)
#pragma unroll
        for (int r = 0; r < 8; ++r) {
            float* crow = C + (size_t)(m_base + 16 * i + r + 8 * half) * N + n_base + col;
#pragma unroll
            for (int j = 0; j < 4; ++j) crow[16 * j] = acc[i][j][r];
        }
}

// ---------------------------------------------------------------- RoPE + scatter
// qkv[ROWS][3072] fp32 -> q_bf[B][H][S][D] (RoPE, pre-scaled by 1/sqrt(HD)),
// k_bf[B][KV][S][D] (RoPE), vt_bf[B][KV][D][S] (transposed).
__global__ __launch_bounds__(256)
void rope_scatter_kernel(const float* __restrict__ qkv,
                         __bf16* __restrict__ q_bf, __bf16* __restrict__ k_bf,
                         __bf16* __restrict__ vt_bf) {
    const int row = blockIdx.x;            // 0..ROWS-1
    const int b = row / SEQ, s = row % SEQ;
    const float* base = qkv + (size_t)row * QKV_N;
    const float LN1E4_OVER_64 = 9.210340371976184f / 64.0f;
    const float qscale = 0.08838834764831845f;           // 1/sqrt(128) folded into Q

    for (int idx = threadIdx.x; idx < NHEADS * HD; idx += blockDim.x) {
        int h = idx >> 7, d = idx & 127;
        float x  = base[h * HD + d];
        float xp = base[h * HD + (d < 64 ? d + 64 : d - 64)];
        float inv = __expf(-(float)(d & 63) * LN1E4_OVER_64);
        float ang = (float)s * inv, sn, cs;
        sincosf(ang, &sn, &cs);
        float rh = (d < 64) ? -xp : xp;
        q_bf[(((size_t)b * NHEADS + h) * SEQ + s) * HD + d] = f2bf((x * cs + rh * sn) * qscale);
    }
    for (int idx = threadIdx.x; idx < NKV * HD; idx += blockDim.x) {
        int kh = idx >> 7, d = idx & 127;
        float x  = base[HID + kh * HD + d];
        float xp = base[HID + kh * HD + (d < 64 ? d + 64 : d - 64)];
        float inv = __expf(-(float)(d & 63) * LN1E4_OVER_64);
        float ang = (float)s * inv, sn, cs;
        sincosf(ang, &sn, &cs);
        float rh = (d < 64) ? -xp : xp;
        k_bf[(((size_t)b * NKV + kh) * SEQ + s) * HD + d] = f2bf(x * cs + rh * sn);
    }
    for (int idx = threadIdx.x; idx < NKV * HD; idx += blockDim.x) {
        int kh = idx >> 7, d = idx & 127;
        vt_bf[(((size_t)b * NKV + kh) * HD + d) * SEQ + s] = f2bf(base[HID + 512 + kh * HD + d]);
    }
}

// ---------------------------------------------------------------- flash attention
// grid (S/64, NHEADS, BATCH), 128 threads = 4 waves; wave owns 16 q-rows.
// K/V blocks staged into LDS (shared by all 4 waves), double-buffered async copies.

// K tile [32][HD] (8KB) and V tile [HD][32] (8KB): 128 threads x 4 chunks of 16B each.
__device__ __forceinline__ void stage_kv(const __bf16* __restrict__ Kb,
                                         const __bf16* __restrict__ Vt, int k0,
                                         __bf16* kt, __bf16* vt) {
    const int t = threadIdx.x;             // 0..127
#pragma unroll
    for (int c = 0; c < 4; ++c) {          // K tile: 512 chunks, 16 per row (256B rows)
        int chunk = t + c * 128;
        int row = chunk >> 4;              // key-in-block 0..31
        int off = (chunk & 15) * 8;        // bf16 offset within row
        cp16_g2l(kt + row * HD + off, Kb + (size_t)(k0 + row) * HD + off);
    }
#pragma unroll
    for (int c = 0; c < 4; ++c) {          // V tile: 512 chunks, 4 per row (64B rows)
        int chunk = t + c * 128;
        int row = chunk >> 2;              // d 0..127
        int off = (chunk & 3) * 8;
        cp16_g2l(vt + row * 32 + off, Vt + (size_t)row * SEQ + k0 + off);
    }
}

__global__ __launch_bounds__(128)
void flash_attn_kernel(const __bf16* __restrict__ q_bf, const __bf16* __restrict__ k_bf,
                       const __bf16* __restrict__ vt_bf, __bf16* __restrict__ o_bf) {
    __shared__ __align__(32) __bf16 ktile[2][32 * HD];   // 2 x 8KB
    __shared__ __align__(32) __bf16 vtile[2][HD * 32];   // 2 x 8KB
    __shared__ __align__(32) __bf16 plds[4 * 16 * 32];   // per-wave P bounce, 4KB

    const int lane = threadIdx.x & 31;
    const int wave = threadIdx.x >> 5;
    const int half = lane >> 4;
    const int col  = lane & 15;
    const int h  = blockIdx.y;
    const int b  = blockIdx.z;
    const int kh = h >> 2;                               // GQA: jnp.repeat(k,4,axis=2)
    const int q0   = blockIdx.x * 64 + wave * 16;
    const int qmax = blockIdx.x * 64 + 63;               // block-uniform causal bound

    const __bf16* Qb = q_bf  + (((size_t)b * NHEADS + h) * SEQ + q0) * HD;  // [16][128]
    const __bf16* Kb = k_bf  + (((size_t)b * NKV + kh) * SEQ) * HD;         // [S][128]
    const __bf16* Vt = vt_bf + (((size_t)b * NKV + kh) * HD) * SEQ;         // [128][S]
    __bf16* pl = plds + wave * 16 * 32;

    v16bf qf[4];                                         // resident Q fragments
#pragma unroll
    for (int dk = 0; dk < 4; ++dk) qf[dk] = load_a_frag(Qb, HD, 0, dk * 32);

    v8f o[8];
#pragma unroll
    for (int j = 0; j < 8; ++j) o[j] = zero8();
    float rowm[8], rowl[8];
#pragma unroll
    for (int r = 0; r < 8; ++r) { rowm[r] = -1e30f; rowl[r] = 0.0f; }

    stage_kv(Kb, Vt, 0, ktile[0], vtile[0]);
    async_join();
    __syncthreads();

    int cur = 0;
    for (int k0 = 0; k0 <= qmax; k0 += 32) {
        if (k0 + 32 <= qmax)                              // prefetch next block (other buffer)
            stage_kv(Kb, Vt, k0 + 32, ktile[cur ^ 1], vtile[cur ^ 1]);

        if (k0 <= q0 + 15) {                              // wave-uniform causal skip
            const __bf16* kt = ktile[cur];
            const __bf16* vtl = vtile[cur];

            // ---- scores: two 16x16 tiles over keys [k0..k0+15], [k0+16..k0+31]
            v8f c0 = zero8(), c1 = zero8();
#pragma unroll
            for (int dk = 0; dk < 4; ++dk) {
                v16bf kb0 = *(const v16bf*)(kt + (size_t)(col)      * HD + dk * 32 + 16 * half);
                v16bf kb1 = *(const v16bf*)(kt + (size_t)(16 + col) * HD + dk * 32 + 16 * half);
                c0 = __builtin_amdgcn_wmma_f32_16x16x32_bf16(false, qf[dk], false, kb0, (short)0, c0, false, false);
                c1 = __builtin_amdgcn_wmma_f32_16x16x32_bf16(false, qf[dk], false, kb1, (short)0, c1, false, false);
            }

            // ---- online softmax (rows r+8*half live entirely within a 16-lane half)
#pragma unroll
            for (int r = 0; r < 8; ++r) {
                const int qrow = q0 + r + 8 * half;
                float s0 = c0[r]; if (k0 + col      > qrow) s0 = -1e30f;
                float s1 = c1[r]; if (k0 + 16 + col > qrow) s1 = -1e30f;
                float ml = fmaxf(s0, s1);
#pragma unroll
                for (int m = 1; m < 16; m <<= 1) ml = fmaxf(ml, __shfl_xor(ml, m, 32));
                float mn = fmaxf(rowm[r], ml);
                float corr = __expf(rowm[r] - mn);
                rowm[r] = mn;
                float p0 = __expf(s0 - mn);
                float p1 = __expf(s1 - mn);
                float ls = p0 + p1;
#pragma unroll
                for (int m = 1; m < 16; m <<= 1) ls += __shfl_xor(ls, m, 32);
                rowl[r] = rowl[r] * corr + ls;
#pragma unroll
                for (int j = 0; j < 8; ++j) o[j][r] *= corr;
                pl[(r + 8 * half) * 32 + col]      = f2bf(p0);
                pl[(r + 8 * half) * 32 + 16 + col] = f2bf(p1);
            }
            __builtin_amdgcn_wave_barrier();             // same-wave DS ops are in-order

            // ---- reload P in A-layout (row = col, K = i + 16*half)
            const __bf16* pa = pl + (size_t)col * 32 + 16 * half;
            v8bf plo = *(const v8bf*)(pa);
            v8bf phi = *(const v8bf*)(pa + 8);
            v16bf paf = __builtin_shufflevector(plo, phi, 0,1,2,3,4,5,6,7,8,9,10,11,12,13,14,15);
            __builtin_amdgcn_wave_barrier();

            // ---- O += P(16x32) * V(32x128): 8 d-chunks of 16
#pragma unroll
            for (int j = 0; j < 8; ++j) {
                v16bf vb = *(const v16bf*)(vtl + (size_t)(j * 16 + col) * 32 + 16 * half);
                o[j] = __builtin_amdgcn_wmma_f32_16x16x32_bf16(false, paf, false, vb, (short)0, o[j], false, false);
            }
        }

        async_join();                                    // our async copies into cur^1 done
        __syncthreads();                                 // everyone done reading cur
        cur ^= 1;
    }

    // ---- normalize + store to [B][S][H*D] bf16
#pragma unroll
    for (int j = 0; j < 8; ++j)
#pragma unroll
        for (int r = 0; r < 8; ++r) {
            const int qrow = q0 + r + 8 * half;
            float val = o[j][r] / rowl[r];
            o_bf[((size_t)b * SEQ + qrow) * HID + h * HD + j * 16 + col] = f2bf(val);
        }
}

// ---------------------------------------------------------------- launch
extern "C" void kernel_launch(void* const* d_in, const int* in_sizes, int n_in,
                              void* d_out, int out_size, void* d_ws, size_t ws_size,
                              hipStream_t stream) {
    const float* hs = (const float*)d_in[0];   // [B][S][2048]
    const float* wq = (const float*)d_in[1];   // [2048][2048]
    const float* wk = (const float*)d_in[2];   // [2048][512]
    const float* wv = (const float*)d_in[3];   // [2048][512]
    const float* wo = (const float*)d_in[4];   // [2048][2048]
    float* out = (float*)d_out;                // [B][S][2048] fp32

    __bf16* hid_bf = (__bf16*)d_ws;                                   // 4096*2048
    __bf16* wqkvt  = hid_bf + (size_t)ROWS * HID;                     // 3072*2048 (K-major)
    __bf16* wot    = wqkvt  + (size_t)QKV_N * HID;                    // 2048*2048 (K-major)
    float*  qkvf   = (float*)(wot + (size_t)HID * HID);               // 4096*3072
    __bf16* q_bf   = (__bf16*)(qkvf + (size_t)ROWS * QKV_N);          // [B][H][S][D]
    __bf16* k_bf   = q_bf  + (size_t)BATCH * NHEADS * SEQ * HD;       // [B][KV][S][D]
    __bf16* vt_bf  = k_bf  + (size_t)BATCH * NKV * SEQ * HD;          // [B][KV][D][S]
    __bf16* att_bf = vt_bf + (size_t)BATCH * NKV * HD * SEQ;          // [B][S][2048]

    // 1) convert hidden, transpose+fuse weights to K-major bf16
    f32_to_bf16_kernel<<<2048, 256, 0, stream>>>(hs, hid_bf, (long long)ROWS * HID);
    transpose_f32_to_bf16_kernel<<<2048, 256, 0, stream>>>(wq, wqkvt,                      HID, HID);
    transpose_f32_to_bf16_kernel<<<1024, 256, 0, stream>>>(wk, wqkvt + (size_t)HID * HID,  HID, 512);
    transpose_f32_to_bf16_kernel<<<1024, 256, 0, stream>>>(wv, wqkvt + (size_t)2560 * HID, HID, 512);
    transpose_f32_to_bf16_kernel<<<2048, 256, 0, stream>>>(wo, wot,                        HID, HID);

    // 2) fused QKV projection: [4096x2048] x [2048x3072]
    gemm_bf16_wmma<<<dim3(QKV_N / 128, ROWS / 128), 128, 0, stream>>>(
        hid_bf, wqkvt, qkvf, ROWS, QKV_N, HID);

    // 3) RoPE + head-major scatter (V transposed, Q pre-scaled)
    rope_scatter_kernel<<<ROWS, 256, 0, stream>>>(qkvf, q_bf, k_bf, vt_bf);

    // 4) causal flash attention (LDS-shared, double-buffered K/V)
    flash_attn_kernel<<<dim3(SEQ / 64, NHEADS, BATCH), 128, 0, stream>>>(
        q_bf, k_bf, vt_bf, att_bf);

    // 5) output projection: [4096x2048] x [2048x2048] -> fp32 out
    gemm_bf16_wmma<<<dim3(HID / 128, ROWS / 128), 128, 0, stream>>>(
        att_bf, wot, out, ROWS, HID, HID);
}
